// MyGNN_22471268892848
// MI455X (gfx1250) — matrix operand order repaired and checked by reading, not compiled, not run on hardware
//
#include <hip/hip_runtime.h>
#include <hip/hip_bf16.h>
#include <math.h>

#define NN 100000   // nodes
#define NE 800000   // edges
#define CH 128      // channels (IN == H == 128)

typedef __attribute__((ext_vector_type(16))) __bf16 bf16x16;
typedef __attribute__((ext_vector_type(8)))  float  f32x8;
typedef __attribute__((ext_vector_type(4)))  unsigned int u32x4;
typedef unsigned short u16;

union FragU {
    bf16x16      bf;
    u32x4        q[2];
    unsigned int w[8];
};

// f32 -> bf16 bits, round-to-nearest-even
__device__ __forceinline__ unsigned bf16b(float x) {
    unsigned u = __float_as_uint(x);
    u += 0x7fffu + ((u >> 16) & 1u);
    return u >> 16;
}
__device__ __forceinline__ unsigned pk2(float lo, float hi) {
    return bf16b(lo) | (bf16b(hi) << 16);
}

// ---- WMMA fragment helpers (ISA 7.12.2 layouts, wave32) -------------------
// A-matrix 16x32 bf16: lane m=lane&15 is the row, hi=lane>>4 selects K half.
// VGPR0-3: K = 8*hi + 2v,2v+1 ; VGPR4-7: K = 16 + 8*hi + 2(v-4),+1
// Rows are bf16 in memory -> fragment = two b128 loads, no conversion VALU.
__device__ __forceinline__ bf16x16 afrag_bf16(const u16* row, int kk, int hi) {
    FragU f;
    f.q[0] = *(const u32x4*)(row + kk + 8 * hi);
    f.q[1] = *(const u32x4*)(row + kk + 16 + 8 * hi);
    return f.bf;
}
__device__ __forceinline__ bf16x16 afrag_zero() {
    FragU f;
    #pragma unroll
    for (int i = 0; i < 8; ++i) f.w[i] = 0u;
    return f.bf;
}
// B-matrix 32x16 bf16 from transposed weights Wt[n][k] (global, bf16 bits):
// lane holds column n; VGPR0-3: K=kb+8*hi..+7, VGPR4-7: K=kb+16+8*hi..+7 (2/VGPR)
__device__ __forceinline__ bf16x16 bfrag(const u16* __restrict__ Wt, int n, int kb, int hi, int K) {
    const u16* p = Wt + (size_t)n * K + kb + 8 * hi;
    FragU f;
    f.q[0] = *(const u32x4*)(p);
    f.q[1] = *(const u32x4*)(p + 16);
    return f.bf;
}
__device__ __forceinline__ f32x8 wmma_bf16(bf16x16 a, bf16x16 b, f32x8 c) {
    return __builtin_amdgcn_wmma_f32_16x16x32_bf16(false, a, false, b, (short)0, c, false, false);
}

// ---- small utility kernels ------------------------------------------------
__global__ void k_zero(float* __restrict__ p, int n) {
    int i = blockIdx.x * 256 + threadIdx.x;
    if (i < n) p[i] = 0.f;
}

__global__ void k_deg(const int* __restrict__ dst, float* __restrict__ deg) {
    int e = blockIdx.x * 256 + threadIdx.x;
    if (e < NE) atomicAdd(&deg[dst[e]], 1.0f);
}

__global__ void k_rsqrt(float* __restrict__ deg) {
    int i = blockIdx.x * 256 + threadIdx.x;
    if (i < NN) deg[i] = rsqrtf(deg[i] + 1.0f);   // +1 self loop; always > 0
}

// f32 -> bf16 bulk convert (float4 in, packed uint2 out); n % 4 == 0
__global__ void k_cvt_bf16(const float* __restrict__ src, u16* __restrict__ dst, int n4) {
    int i = blockIdx.x * 256 + threadIdx.x;
    if (i < n4) {
        float4 v = ((const float4*)src)[i];
        ((uint2*)dst)[i] = make_uint2(pk2(v.x, v.y), pk2(v.z, v.w));
    }
}

// bf16-transpose all four weight matrices into workspace: Wt[n*K + k]
__global__ void k_prep_w(const float* __restrict__ W1, const float* __restrict__ W2,
                         const float* __restrict__ Wm1, const float* __restrict__ Wm2,
                         u16* __restrict__ W1t, u16* __restrict__ W2t,
                         u16* __restrict__ Wm1t, u16* __restrict__ Wm2t) {
    int i = blockIdx.x * 256 + threadIdx.x;
    if (i < 16384) {                       // W1 [128][128]
        int k = i >> 7, n = i & 127;
        W1t[n * 128 + k] = (u16)bf16b(W1[i]);
    } else if (i < 32768) {                // W2 [128][128]
        int j = i - 16384; int k = j >> 7, n = j & 127;
        W2t[n * 128 + k] = (u16)bf16b(W2[j]);
    } else if (i < 65536) {                // Wm1 [256][128]
        int j = i - 32768; int k = j >> 7, n = j & 127;
        Wm1t[n * 256 + k] = (u16)bf16b(Wm1[j]);
    } else if (i < 81920) {                // Wm2 [128][128]
        int j = i - 65536; int k = j >> 7, n = j & 127;
        Wm2t[n * 128 + k] = (u16)bf16b(Wm2[j]);
    }
}

// ---- node GEMM: C[N,128] = Abf[N,128](bf16) @ W  (WMMA bf16, f32 acc) -----
__global__ __launch_bounds__(256) void k_gemm_node(const u16* __restrict__ Abf,
                                                   const u16* __restrict__ Wt,
                                                   float* __restrict__ C, int nrows) {
    const int lane = threadIdx.x & 31, w = threadIdx.x >> 5;
    const int m16 = lane & 15, hi = lane >> 4;
    const int rowBase = blockIdx.x * 128 + w * 16;
    const int row = rowBase + m16;
    const bool valid = row < nrows;
    const u16* rowp = Abf + (size_t)row * CH;

    f32x8 acc[8];
    #pragma unroll
    for (int t = 0; t < 8; ++t) acc[t] = (f32x8){0.f, 0.f, 0.f, 0.f, 0.f, 0.f, 0.f, 0.f};

    #pragma unroll
    for (int ks = 0; ks < 4; ++ks) {
        const int kb = ks * 32;
        bf16x16 a = valid ? afrag_bf16(rowp, kb, hi) : afrag_zero();
        #pragma unroll
        for (int t = 0; t < 8; ++t) {
            bf16x16 b = bfrag(Wt, t * 16 + m16, kb, hi, 128);
            acc[t] = wmma_bf16(a, b, acc[t]);
        }
    }
    // D layout: VGPR g -> row m = g + 8*hi, col n = t*16 + m16
    #pragma unroll
    for (int t = 0; t < 8; ++t)
        #pragma unroll
        for (int g = 0; g < 8; ++g) {
            int r = rowBase + 8 * hi + g;
            if (r < nrows) C[(size_t)r * CH + t * 16 + m16] = acc[t][g];
        }
}

// ---- edge scatter: agg[d] += h[s] * dinv[s]*dinv[d]  (32 threads / edge) --
__global__ __launch_bounds__(256) void k_scatter(const float* __restrict__ h,
                                                 const int* __restrict__ ei,
                                                 const float* __restrict__ dinv,
                                                 float* __restrict__ agg) {
    int t = blockIdx.x * 256 + threadIdx.x;
    int e = t >> 5;
    int c = (t & 31) << 2;
    if (e >= NE) return;
    int s = ei[e], d = ei[NE + e];
    float wgt = dinv[s] * dinv[d];
    float4 v = *(const float4*)(h + (size_t)s * CH + c);
    float* ap = agg + (size_t)d * CH + c;
    atomicAdd(ap + 0, v.x * wgt);
    atomicAdd(ap + 1, v.y * wgt);
    atomicAdd(ap + 2, v.z * wgt);
    atomicAdd(ap + 3, v.w * wgt);
}

// ---- finalize: h = relu(agg + h*dinv^2 + b); also writes bf16 mirror ------
__global__ void k_finalize(float* __restrict__ h, const float* __restrict__ agg,
                           const float* __restrict__ dinv, const float* __restrict__ b,
                           u16* __restrict__ hbf) {
    int i = blockIdx.x * 256 + threadIdx.x;
    if (i < NN * CH) {
        int r = i >> 7, c = i & 127;
        float di = dinv[r];
        float v = fmaxf(agg[i] + h[i] * di * di + b[c], 0.f);
        h[i] = v;
        hbf[i] = (u16)bf16b(v);
    }
}

// ---- fused edge MLP: sigmoid(relu(relu([h_s|h_d]Wm1+b)Wm2+b)Wm3+b) --------
// A-side comes straight from the bf16 mirror of h (L2-resident, 26 MB).
__global__ __launch_bounds__(256) void k_edge_mlp(const u16* __restrict__ hbf,
                                                  const int* __restrict__ ei,
                                                  const u16* __restrict__ Wm1t,
                                                  const u16* __restrict__ Wm2t,
                                                  const float* __restrict__ Wm3,
                                                  const float* __restrict__ bm1,
                                                  const float* __restrict__ bm2,
                                                  const float* __restrict__ bm3,
                                                  float* __restrict__ out) {
    __shared__ u16 z1buf[8][16 * 136];     // per-wave z1 tile, padded stride
    const int lane = threadIdx.x & 31, w = threadIdx.x >> 5;
    const int m16 = lane & 15, hi = lane >> 4;
    const int eBase = blockIdx.x * 128 + w * 16;   // E = 6250*128 exactly
    const int e = eBase + m16;
    const int s = ei[e], d = ei[NE + e];
    const u16* srow = hbf + (size_t)s * CH;
    const u16* drow = hbf + (size_t)d * CH;
    __builtin_prefetch(drow, 0, 0);        // global_prefetch_b8: warm dst row

    // ---- layer 1: [16 x 256] @ Wm1 -> z1 [16 x 128]
    f32x8 acc[8];
    #pragma unroll
    for (int t = 0; t < 8; ++t) acc[t] = (f32x8){0.f, 0.f, 0.f, 0.f, 0.f, 0.f, 0.f, 0.f};
    #pragma unroll
    for (int ks = 0; ks < 8; ++ks) {
        const int kb = ks * 32;
        const u16* rp = (kb < 128) ? srow : drow;     // ef = [h_src | h_dst]
        bf16x16 a = afrag_bf16(rp, kb & 127, hi);
        #pragma unroll
        for (int t = 0; t < 8; ++t) {
            bf16x16 b = bfrag(Wm1t, t * 16 + m16, kb, hi, 256);
            acc[t] = wmma_bf16(a, b, acc[t]);
        }
    }
    // bias+relu, re-layout D->A via per-wave LDS tile (bf16)
    u16* zb = z1buf[w];
    #pragma unroll
    for (int t = 0; t < 8; ++t) {
        float bb = bm1[t * 16 + m16];
        #pragma unroll
        for (int g = 0; g < 8; ++g) {
            float v = fmaxf(acc[t][g] + bb, 0.f);
            zb[(g + 8 * hi) * 136 + t * 16 + m16] = (u16)bf16b(v);
        }
    }

    // ---- layer 2: z1 [16 x 128] @ Wm2 -> z2 [16 x 128]
    f32x8 acc2[8];
    #pragma unroll
    for (int t = 0; t < 8; ++t) acc2[t] = (f32x8){0.f, 0.f, 0.f, 0.f, 0.f, 0.f, 0.f, 0.f};
    #pragma unroll
    for (int ks = 0; ks < 4; ++ks) {
        const int kb = ks * 32;
        FragU fa;
        fa.q[0] = *(const u32x4*)(zb + m16 * 136 + kb + 8 * hi);
        fa.q[1] = *(const u32x4*)(zb + m16 * 136 + kb + 16 + 8 * hi);
        #pragma unroll
        for (int t = 0; t < 8; ++t) {
            bf16x16 b = bfrag(Wm2t, t * 16 + m16, kb, hi, 128);
            acc2[t] = wmma_bf16(fa.bf, b, acc2[t]);
        }
    }

    // ---- layer 3: relu(z2 + bm2) . Wm3 per edge, 16-lane shuffle reduce
    float p[8];
    #pragma unroll
    for (int g = 0; g < 8; ++g) p[g] = 0.f;
    #pragma unroll
    for (int t = 0; t < 8; ++t) {
        float bb = bm2[t * 16 + m16];
        float ww = Wm3[t * 16 + m16];
        #pragma unroll
        for (int g = 0; g < 8; ++g)
            p[g] += fmaxf(acc2[t][g] + bb, 0.f) * ww;
    }
    #pragma unroll
    for (int g = 0; g < 8; ++g) {
        #pragma unroll
        for (int off = 8; off > 0; off >>= 1)
            p[g] += __shfl_xor(p[g], off, 16);     // reduce within 16-lane half
    }
    if (m16 == 0) {
        float b3 = bm3[0];
        #pragma unroll
        for (int g = 0; g < 8; ++g) {
            int ee = eBase + 8 * hi + g;           // hi=0: edges 0-7, hi=1: 8-15
            out[ee] = 1.f / (1.f + expf(-(p[g] + b3)));
        }
    }
}

// ---------------------------------------------------------------------------
extern "C" void kernel_launch(void* const* d_in, const int* in_sizes, int n_in,
                              void* d_out, int out_size, void* d_ws, size_t ws_size,
                              hipStream_t stream) {
    (void)in_sizes; (void)n_in; (void)out_size; (void)ws_size;
    const float* x   = (const float*)d_in[0];
    const int*   ei  = (const int*)d_in[1];
    const float* W1  = (const float*)d_in[2];
    const float* b1  = (const float*)d_in[3];
    const float* W2  = (const float*)d_in[4];
    const float* b2  = (const float*)d_in[5];
    const float* Wm1 = (const float*)d_in[6];
    const float* bm1 = (const float*)d_in[7];
    const float* Wm2 = (const float*)d_in[8];
    const float* bm2 = (const float*)d_in[9];
    const float* Wm3 = (const float*)d_in[10];
    const float* bm3 = (const float*)d_in[11];
    float* out = (float*)d_out;

    // workspace carve (256B aligned)
    char* p = (char*)d_ws;
    auto take = [&](size_t bytes) { char* r = p; p += (bytes + 255) & ~(size_t)255; return r; };
    float* dinv = (float*)take((size_t)NN * 4);
    float* hA   = (float*)take((size_t)NN * CH * 4);
    float* hB   = (float*)take((size_t)NN * CH * 4);
    float* agg  = (float*)take((size_t)NN * CH * 4);
    u16*   hbf  = (u16*)take((size_t)NN * CH * 2);   // bf16 mirror of current h
    u16* W1t  = (u16*)take(128 * 128 * 2);
    u16* W2t  = (u16*)take(128 * 128 * 2);
    u16* Wm1t = (u16*)take(256 * 128 * 2);
    u16* Wm2t = (u16*)take(128 * 128 * 2);

    const int nc = NN * CH;

    // degrees + symmetric norm
    k_zero<<<(NN + 255) / 256, 256, 0, stream>>>(dinv, NN);
    k_deg<<<(NE + 255) / 256, 256, 0, stream>>>(ei + NE, dinv);
    k_rsqrt<<<(NN + 255) / 256, 256, 0, stream>>>(dinv);

    // bf16 transposed weights + bf16 input features
    k_prep_w<<<320, 256, 0, stream>>>(W1, W2, Wm1, Wm2, W1t, W2t, Wm1t, Wm2t);
    k_cvt_bf16<<<(nc / 4 + 255) / 256, 256, 0, stream>>>(x, hbf, nc / 4);

    // GCN layer 1 (A = bf16(x))
    k_gemm_node<<<(NN + 127) / 128, 256, 0, stream>>>(hbf, W1t, hA, NN);
    k_zero<<<(nc + 255) / 256, 256, 0, stream>>>(agg, nc);
    k_scatter<<<(NE * 32) / 256, 256, 0, stream>>>(hA, ei, dinv, agg);
    k_finalize<<<(nc + 255) / 256, 256, 0, stream>>>(hA, agg, dinv, b1, hbf);

    // GCN layer 2 (A = bf16(h1))
    k_gemm_node<<<(NN + 127) / 128, 256, 0, stream>>>(hbf, W2t, hB, NN);
    k_zero<<<(nc + 255) / 256, 256, 0, stream>>>(agg, nc);
    k_scatter<<<(NE * 32) / 256, 256, 0, stream>>>(hB, ei, dinv, agg);
    k_finalize<<<(nc + 255) / 256, 256, 0, stream>>>(hB, agg, dinv, b2, hbf);

    // fused edge-scoring MLP (gathers bf16 rows straight from L2)
    k_edge_mlp<<<NE / 128, 256, 0, stream>>>(hbf, ei, Wm1t, Wm2t, Wm3, bm1, bm2, bm3, out);
}
